// HungarianMatcher_56212531970180
// MI455X (gfx1250) — compile-verified, matrix operand order checked by reading, not run on hardware
//
#include <hip/hip_runtime.h>
#include <hip/hip_bf16.h>

typedef __attribute__((ext_vector_type(16))) _Float16     v16h;
typedef __attribute__((ext_vector_type(8)))  float        v8f;
typedef __attribute__((ext_vector_type(8)))  unsigned int v8u;

#define NFRM 4
#define BB   64
#define NQ   300
#define NCL  512
#define NT   50
#define KK   34
#define QT   19   // ceil(300/16)
#define TT   4    // ceil(50/16)

// ---------------------------------------------------------------------------
// Kernel 1: softmax row statistics (max, 1/sum(exp)) — one wave32 per query row
// ---------------------------------------------------------------------------
__global__ __launch_bounds__(256) void softmax_stats(
    const float* __restrict__ logits, const int* __restrict__ framep,
    float* __restrict__ rowmax, float* __restrict__ rowinv) {
  const int frame = *framep;
  const int wave  = (int)((blockIdx.x * blockDim.x + threadIdx.x) >> 5);
  const int lane  = (int)(threadIdx.x & 31);
  if (wave >= BB * NQ) return;
  const float* row = logits + ((size_t)frame * BB * NQ + wave) * NCL;
  float x[16];
#pragma unroll
  for (int s = 0; s < 16; ++s) x[s] = row[lane + 32 * s];   // coalesced 128B/step
  float mx = x[0];
#pragma unroll
  for (int s = 1; s < 16; ++s) mx = fmaxf(mx, x[s]);
#pragma unroll
  for (int m = 16; m >= 1; m >>= 1) mx = fmaxf(mx, __shfl_xor(mx, m, 32));
  float sum = 0.f;
#pragma unroll
  for (int s = 0; s < 16; ++s) sum += __expf(x[s] - mx);
#pragma unroll
  for (int m = 16; m >= 1; m >>= 1) sum += __shfl_xor(sum, m, 32);
  if (lane == 0) { rowmax[wave] = mx; rowinv[wave] = 1.0f / sum; }
}

// ---------------------------------------------------------------------------
// Kernel 2: cost matrix via WMMA. cls_cost as GEMM against one-hot(tgt_cls):
//   D[q,t] = sum_c exp(logit[q,c]-max[q]) * onehot[c,t];  cls = -D * inv[q]
// One wave32 per (b, 16-query tile): A fragment (exp of logits) built once per
// K-slab and multiplied against 4 one-hot B fragments (all 50 targets) with
// 4 back-to-back v_wmma into 4 accumulators. One-hot patterns are precomputed
// per lane (hit slab + 8-dword mask); in-loop B build is 1 cmp + 8 cndmask.
// Writes cost transposed [b][t][q] (rows = targets) for the LSA stage.
// ---------------------------------------------------------------------------
__global__ __launch_bounds__(32) void wmma_cost(
    const float* __restrict__ logits, const float* __restrict__ pkpt,
    const float* __restrict__ tkpt,  const int* __restrict__ tcls,
    const int* __restrict__ framep,
    const float* __restrict__ rowmax, const float* __restrict__ rowinv,
    float* __restrict__ cost) {
  const int frame = *framep;
  const int lane  = (int)threadIdx.x;          // 0..31, full wave, EXEC all ones
  const int bid   = (int)blockIdx.x;
  const int b     = bid / QT;
  const int qt    = bid % QT;
  const int h     = lane >> 4;                 // half-wave: K interleave select
  const int m     = lane & 15;

  const int q  = qt * 16 + m;                  // A-fragment row (M = m)
  const int qc = (q < NQ) ? q : (NQ - 1);      // clamp; masked at store

  // Per target tile: class id (B-fragment column N = m), hit slab, and the
  // one-hot dword pattern for this lane's K-interleave slots.
  // Lane half h covers in-slab offsets [8h,8h+8) (elems 0..7) and
  // [16+8h,24+8h) (elems 8..15); elem -> dword elem>>1, halfword elem&1.
  int hitslab[TT];
  v8u pre[TT];
#pragma unroll
  for (int tt = 0; tt < TT; ++tt) {
    const int t = tt * 16 + m;
    const int d = (t < NT) ? tcls[b * NT + t] : -1;   // -1 => never matches
    hitslab[tt] = d >> 5;                              // slab index or -1
    const int o  = d & 31;                             // offset within slab
    int elem = -1;
    if (d >= 0) {
      if (o >= 8 * h && o < 8 * h + 8)            elem = o - 8 * h;
      else if (o >= 16 + 8 * h && o < 24 + 8 * h) elem = o - 16 - 8 * h + 8;
    }
    v8u p = {};
#pragma unroll
    for (int r = 0; r < 8; ++r)
      p[r] = ((elem >> 1) == r) ? (0x3C00u << ((elem & 1) * 16)) : 0u;
    pre[tt] = p;
  }

  const float* Lrow = logits + (((size_t)frame * BB + b) * NQ + qc) * NCL;
  const float  mx   = rowmax[b * NQ + qc];
  const v8u    vz   = {};

  v8f acc0 = {}, acc1 = {}, acc2 = {}, acc3 = {};
#pragma unroll 2
  for (int kb = 0; kb < NCL; kb += 32) {
    if (kb + 32 < NCL) __builtin_prefetch(Lrow + kb + 32, 0, 1);
    const int slab = kb >> 5;
    const int c0 = kb + 8 * h;        // A lane holds K {c0..c0+7} and {c1..c1+7}
    const int c1 = kb + 16 + 8 * h;
    v16h a;
#pragma unroll
    for (int e = 0; e < 8; ++e) {
      a[e]     = (_Float16)__expf(Lrow[c0 + e] - mx);
      a[e + 8] = (_Float16)__expf(Lrow[c1 + e] - mx);
    }
    const v16h bh0 = __builtin_bit_cast(v16h, (hitslab[0] == slab) ? pre[0] : vz);
    const v16h bh1 = __builtin_bit_cast(v16h, (hitslab[1] == slab) ? pre[1] : vz);
    const v16h bh2 = __builtin_bit_cast(v16h, (hitslab[2] == slab) ? pre[2] : vz);
    const v16h bh3 = __builtin_bit_cast(v16h, (hitslab[3] == slab) ? pre[3] : vz);
    acc0 = __builtin_amdgcn_wmma_f32_16x16x32_f16(false, a, false, bh0,
                                                  (short)0, acc0, false, false);
    acc1 = __builtin_amdgcn_wmma_f32_16x16x32_f16(false, a, false, bh1,
                                                  (short)0, acc1, false, false);
    acc2 = __builtin_amdgcn_wmma_f32_16x16x32_f16(false, a, false, bh2,
                                                  (short)0, acc2, false, false);
    acc3 = __builtin_amdgcn_wmma_f32_16x16x32_f16(false, a, false, bh3,
                                                  (short)0, acc3, false, false);
  }

  // D layout: lane holds N = m (target), M = v + 8*h (query) for v in 0..7
  const float* pbase = pkpt + ((size_t)frame * BB + b) * NQ * KK;
  const float* tbase = tkpt + (size_t)b * NT * KK;
#pragma unroll
  for (int v = 0; v < 8; ++v) {
    const int  qq  = qt * 16 + v + 8 * h;
    const bool qok = (qq < NQ);
    const int  qqc = qok ? qq : (NQ - 1);
    const float invq = rowinv[b * NQ + qqc];
    const float* prow = pbase + (size_t)qqc * KK;
    float pr[KK];
#pragma unroll
    for (int k = 0; k < KK; ++k) pr[k] = prow[k];   // reused across 4 t-tiles
    const float cls[TT] = { acc0[v], acc1[v], acc2[v], acc3[v] };
#pragma unroll
    for (int tt = 0; tt < TT; ++tt) {
      const int t = tt * 16 + m;
      if (qok && t < NT) {
        const float* trow = tbase + (size_t)t * KK;
        float kc = 0.f;
#pragma unroll
        for (int k = 0; k < KK; ++k) kc += fabsf(pr[k] - trow[k]);
        cost[((size_t)b * NT + t) * NQ + qq] = kc - cls[tt] * invq; // W=1,1
      }
    }
  }
}

// ---------------------------------------------------------------------------
// Kernel 3: Jonker-Volgenant LSA, one wave32 per batch element.
// Whole 50x300 cost tile + algorithm state resident in LDS (64.2 KB).
// Wave-synchronous Dijkstra; argmin with unassigned-column tie preference
// replicated via packed (pref<<9 | j) key in a 5-step shuffle reduction.
// ---------------------------------------------------------------------------
__global__ __launch_bounds__(32) void lsa_kernel(
    const float* __restrict__ costg, int* __restrict__ out) {
  __shared__ float sC[NT * NQ];      // 60000 B
  __shared__ float sV[NQ];           // duals (cols)
  __shared__ float sShort[NQ];
  __shared__ float sU[NT];           // duals (rows)
  __shared__ short sPath[NQ];
  __shared__ short sR4C[NQ];         // row4col
  __shared__ short sC4R[NT];         // col4row
  __shared__ unsigned char sSC[NQ];
  __shared__ unsigned char sSR[NT];

  const int b    = (int)blockIdx.x;
  const int lane = (int)threadIdx.x;

  // stage cost into LDS (float4, 16B aligned: 60000 % 16 == 0)
  const float4* src  = (const float4*)(costg + (size_t)b * NT * NQ);
  float4*       dst4 = (float4*)sC;
  for (int i = lane; i < (NT * NQ) / 4; i += 32) dst4[i] = src[i];
  for (int j = lane; j < NQ; j += 32) { sV[j] = 0.f; sR4C[j] = -1; }
  for (int i = lane; i < NT; i += 32) { sU[i] = 0.f; sC4R[i] = -1; }
  __syncthreads();

  for (int cur = 0; cur < NT; ++cur) {
    for (int j = lane; j < NQ; j += 32) {
      sShort[j] = __builtin_inff(); sPath[j] = -1; sSC[j] = 0;
    }
    for (int i = lane; i < NT; i += 32) sSR[i] = 0;
    __syncthreads();

    int   i      = cur;
    float minval = 0.f;
    int   sink   = -1;
    while (sink < 0) {
      if (lane == 0) sSR[i] = 1;
      const float ui = sU[i];
      for (int j = lane; j < NQ; j += 32) {
        if (!sSC[j]) {
          const float r = minval + sC[i * NQ + j] - ui - sV[j];
          if (r < sShort[j]) { sShort[j] = r; sPath[j] = (short)i; }
        }
      }
      // local best over owned columns, key = (assigned?1:0)<<9 | j
      float best = __builtin_inff();
      int   bkey = 0x7fffffff;
      for (int j = lane; j < NQ; j += 32) {
        if (!sSC[j]) {
          const float s   = sShort[j];
          const int   key = (((sR4C[j] < 0) ? 0 : 1) << 9) | j;
          if (s < best || (s == best && key < bkey)) { best = s; bkey = key; }
        }
      }
#pragma unroll
      for (int mm = 16; mm >= 1; mm >>= 1) {
        const float ob = __shfl_xor(best, mm, 32);
        const int   ok = __shfl_xor(bkey, mm, 32);
        if (ob < best || (ob == best && ok < bkey)) { best = ob; bkey = ok; }
      }
      const int j = bkey & 511;
      minval = best;
      __syncthreads();
      if (lane == 0) sSC[j] = 1;
      __syncthreads();
      const short rj = sR4C[j];
      if (rj < 0) sink = j; else i = (int)rj;
      __syncthreads();
    }

    // dual updates
    if (lane == 0) sU[cur] += minval;
    for (int ii = lane; ii < NT; ii += 32)
      if (ii != cur && sSR[ii]) sU[ii] += minval - sShort[sC4R[ii]];
    for (int j = lane; j < NQ; j += 32)
      if (sSC[j]) sV[j] -= minval - sShort[j];
    __syncthreads();

    // augment along alternating path (serial pointer chase, lane 0)
    if (lane == 0) {
      int j = sink;
      for (;;) {
        const int i2 = (int)sPath[j];
        sR4C[j] = (short)i2;
        const int nj = (int)sC4R[i2];
        sC4R[i2] = (short)j;
        if (i2 == cur) break;
        j = nj;
      }
    }
    __syncthreads();
  }

  // argsort by matched query index (values distinct => rank is a permutation)
  for (int tt_ = lane; tt_ < NT; tt_ += 32) {
    const int cv = (int)sC4R[tt_];
    int rank = 0;
    for (int s = 0; s < NT; ++s) rank += ((int)sC4R[s] < cv) ? 1 : 0;
    out[b * 2 * NT + rank]      = cv;    // idx_i: query indices ascending
    out[b * 2 * NT + NT + rank] = tt_;   // idx_j: matched target indices
  }
}

// ---------------------------------------------------------------------------
extern "C" void kernel_launch(void* const* d_in, const int* in_sizes, int n_in,
                              void* d_out, int out_size, void* d_ws, size_t ws_size,
                              hipStream_t stream) {
  (void)in_sizes; (void)n_in; (void)out_size; (void)ws_size;
  const float* logits = (const float*)d_in[0];   // [4,64,300,512] f32
  const float* pkpt   = (const float*)d_in[1];   // [4,64,300,34]  f32
  const float* tkpt   = (const float*)d_in[2];   // [64,50,34]     f32
  const int*   tcls   = (const int*)d_in[3];     // [64,50]        i32
  const int*   framep = (const int*)d_in[4];     // scalar         i32

  float* ws     = (float*)d_ws;
  float* cost   = ws;                            // 64*50*300 = 960000 f32
  float* rowmax = ws + (size_t)BB * NT * NQ;     // 19200 f32
  float* rowinv = rowmax + (size_t)BB * NQ;      // 19200 f32

  softmax_stats<<<(BB * NQ + 7) / 8, 256, 0, stream>>>(logits, framep, rowmax, rowinv);
  wmma_cost<<<BB * QT, 32, 0, stream>>>(logits, pkpt, tkpt, tcls, framep,
                                        rowmax, rowinv, cost);
  lsa_kernel<<<BB, 32, 0, stream>>>(cost, (int*)d_out);
}
